// RNNModel_17025250361864
// MI455X (gfx1250) — matrix operand order repaired and checked by reading, not compiled, not run on hardware
//
#include <hip/hip_runtime.h>

// ---------------------------------------------------------------------------
// 2-layer LSTM (B=64, T=400, E=H=500) + MLP head, for MI455X (gfx1250).
// bf16 WMMA (v_wmma_f32_16x16x32_bf16) with fp32 accumulation; fp32 gate math.
// Weights repacked once into the per-lane WMMA B-fragment layout, then each
// step-kernel block DMAs its per-k-step weight slab into dynamic LDS with
// global_load_async_to_lds_b128 (ASYNCcnt, double-buffered) and feeds the
// WMMAs from explicit ds_load_b128 (+ s_wait_dscnt). Both sides of the LDS
// traffic are inline asm on raw LDS offsets so the compiler cannot eliminate
// the pipeline. Recurrence sequenced by stream-ordered launches.
// ---------------------------------------------------------------------------

typedef __attribute__((ext_vector_type(16))) __bf16 v16bf;
typedef __attribute__((ext_vector_type(8)))  float  v8f;

union Frag32 {           // one WMMA 16-bf16 fragment = 32 bytes
    uint4  u[2];
    v16bf  v;
    __bf16 h[16];
};

#define TSTEPS 400
#define BATCH  64
#define EMB    500
#define HID    500
#define NP     512      // padded per-gate column count
#define CT     32       // column tiles (NP/16)
#define KS     32       // K steps (Kp=1024 / 32)
#define LDS_BYTES 8192  // 2 x 4 KB weight slab double buffer (dynamic LDS)
// Kp = 1024: x part in [0,512), h part in [512,1024), both zero-padded.

__device__ __forceinline__ float sigmf(float x) { return 1.0f / (1.0f + __expf(-x)); }

// --- embed[X] -> padded bf16 [T][64][512] -----------------------------------
__global__ void embed_gather_kernel(const int* __restrict__ X,
                                    const float* __restrict__ embed,
                                    __bf16* __restrict__ Xemb)
{
    int idx = blockIdx.x * blockDim.x + threadIdx.x;
    if (idx >= TSTEPS * BATCH * NP) return;
    const int e = idx & 511;
    const int b = (idx >> 9) & 63;
    const int t = idx >> 15;
    float v = 0.0f;
    if (e < EMB) {
        const int tok = X[b * TSTEPS + t];      // X is [B,T]
        v = embed[(size_t)tok * EMB + e];
    }
    Xemb[idx] = (__bf16)v;
}

// --- repack fp32 kernel [1000,2000] into WMMA B-fragment bf16 layout --------
// layout: [ct][gate][ks][lane][16 bf16]; per-lane B fragment:
//   n  = ct*16 + (lane & 15)
//   kp = ks*32 + (lane<16 ? 0 : 16) + e       (e = 0..15)
// kp -> original row k: kp<500 -> kp (x rows); 512<=kp<1012 -> kp-512+500 (h rows)
__global__ void repack_kernel(const float* __restrict__ Kmat,
                              __bf16* __restrict__ Wp)
{
    int idx = blockIdx.x * blockDim.x + threadIdx.x;
    if (idx >= CT * 4 * KS * 32) return;
    const int lane = idx & 31;
    const int ks   = (idx >> 5) & 31;
    const int g    = (idx >> 10) & 3;
    const int ct   = idx >> 12;
    const int n    = (ct << 4) + (lane & 15);
    const int kpb  = (ks << 5) + ((lane & 16) ? 16 : 0);
    Frag32 f;
    #pragma unroll
    for (int e = 0; e < 16; ++e) {
        const int kp = kpb + e;
        float v = 0.0f;
        int k = -1;
        if (kp < EMB)                         k = kp;                  // x rows
        else if (kp >= 512 && kp < 512 + HID) k = (kp - 512) + EMB;    // h rows
        if (k >= 0 && n < HID) v = Kmat[(size_t)k * (4 * HID) + g * HID + n];
        f.h[e] = (__bf16)v;
    }
    uint4* dst = (uint4*)(Wp + (size_t)idx * 16);
    dst[0] = f.u[0];
    dst[1] = f.u[1];
}

// --- async DMA of one k-step weight slab (4 KB) into an LDS buffer ----------
// Global slab layout (per block): gate stride 32768 B, ks stride 1024 B.
// LDS slab layout: [gate][lane][16 bf16] contiguous (4 KB) at offset bufi*4096
// (dynamic LDS starts at workgroup LDS offset 0).
__device__ __forceinline__ void issue_slab_copy(const __bf16* wblk, int ks, int bufi, int tid)
{
    #pragma unroll
    for (int r = 0; r < 2; ++r) {
        const int idx = tid + (r << 7);                 // 0..255 (16B chunks)
        const int g   = idx >> 6;                       // gate
        const int c   = idx & 63;                       // 16B chunk within gate KB
        const unsigned long gaddr = (unsigned long)(size_t)wblk
                                  + (unsigned long)g * 32768ul
                                  + (unsigned long)ks * 1024ul
                                  + (unsigned long)c * 16ul;
        const unsigned laddr = (unsigned)(bufi * 4096 + idx * 16);
        asm volatile("global_load_async_to_lds_b128 %0, %1, off"
                     :: "v"(laddr), "v"(gaddr) : "memory");
    }
}

// --- one LSTM timestep: z = [x,h] @ W + b, fused cell update ----------------
// grid: CT blocks of 128 threads (4 waves). Wave w owns batch rows [16w,16w+16)
// and columns [16*ct, 16*ct+16) of every gate (i,j,f,o). B fragments are
// async-DMA'd to LDS once per block (double-buffered) and shared by all waves.
__global__ void __launch_bounds__(128)
lstm_step_kernel(const __bf16* __restrict__ Ax,     // x part   [64][512] bf16
                 const __bf16* __restrict__ Hprev,  // h_{t-1}  [64][512] bf16
                 const __bf16* __restrict__ Wp,     // repacked weights
                 const float*  __restrict__ bias,   // [2000], gate order i,j,f,o
                 float*        __restrict__ Cst,    // c state  [64][512] f32 (in place)
                 __bf16*       __restrict__ Hout,   // h_t      [64][512] bf16 (other buffer)
                 float*        __restrict__ HoutF)  // optional f32 h_t (layer 2)
{
    extern __shared__ char dyn_lds[];               // 8 KB dynamic LDS (raw offsets)
    (void)dyn_lds;

    const int ct   = blockIdx.x;
    const int tid  = threadIdx.x;
    const int wv   = tid >> 5;
    const int lane = tid & 31;
    const int hsel = (lane & 16) ? 16 : 0;          // A-layout lane-half K offset
    const int arow = (wv << 4) + (lane & 15);       // batch row for A fragment

    const __bf16* wblk = Wp + (size_t)ct * 4 * KS * 32 * 16;   // block's weights

    v8f a0 = {}, a1 = {}, a2 = {}, a3 = {};

    issue_slab_copy(wblk, 0, 0, tid);               // preload slab 0

    #pragma unroll 2
    for (int ks = 0; ks < KS; ++ks) {
        // my slab-ks async ops done; barrier => whole slab present in LDS,
        // and everyone's reads of the other buffer (iteration ks-1) are done.
        asm volatile("s_wait_asynccnt 0x0" ::: "memory");
        __syncthreads();
        if (ks + 1 < KS) issue_slab_copy(wblk, ks + 1, (ks + 1) & 1, tid);

        // A fragment: two 8-elem runs; 8-aligned, never cross x/h split at 512
        const int kb  = ks << 5;
        const int kp0 = kb + hsel;
        const int kp1 = kb + hsel + 8;
        const __bf16* s0 = (kp0 < 512) ? (Ax + arow * 512 + kp0)
                                       : (Hprev + arow * 512 + (kp0 - 512));
        const __bf16* s1 = (kp1 < 512) ? (Ax + arow * 512 + kp1)
                                       : (Hprev + arow * 512 + (kp1 - 512));
        Frag32 af;
        af.u[0] = *(const uint4*)s0;
        af.u[1] = *(const uint4*)s1;

        // B fragments: 8x ds_load_b128 from the current LDS slab, completed by
        // s_wait_dscnt inside the same asm so the WMMAs depend on ready data.
        const unsigned lbase = (unsigned)(((ks & 1) << 12) + lane * 32);
        Frag32 b0, b1, b2, b3;
        asm volatile(
            "ds_load_b128 %0, %8\n\t"
            "ds_load_b128 %1, %8 offset:16\n\t"
            "ds_load_b128 %2, %8 offset:1024\n\t"
            "ds_load_b128 %3, %8 offset:1040\n\t"
            "ds_load_b128 %4, %8 offset:2048\n\t"
            "ds_load_b128 %5, %8 offset:2064\n\t"
            "ds_load_b128 %6, %8 offset:3072\n\t"
            "ds_load_b128 %7, %8 offset:3088\n\t"
            "s_wait_dscnt 0x0"
            : "=v"(b0.u[0]), "=v"(b0.u[1]), "=v"(b1.u[0]), "=v"(b1.u[1]),
              "=v"(b2.u[0]), "=v"(b2.u[1]), "=v"(b3.u[0]), "=v"(b3.u[1])
            : "v"(lbase));

        a0 = __builtin_amdgcn_wmma_f32_16x16x32_bf16(false, af.v, false, b0.v, (short)0, a0, false, false);
        a1 = __builtin_amdgcn_wmma_f32_16x16x32_bf16(false, af.v, false, b1.v, (short)0, a1, false, false);
        a2 = __builtin_amdgcn_wmma_f32_16x16x32_bf16(false, af.v, false, b2.v, (short)0, a2, false, false);
        a3 = __builtin_amdgcn_wmma_f32_16x16x32_bf16(false, af.v, false, b3.v, (short)0, a3, false, false);
    }

    // epilogue: C/D layout -> (M = r + (lane>=16 ? 8:0), N = lane&15)
    const int n = (ct << 4) + (lane & 15);
    if (n < HID) {
        const int moff = (lane & 16) ? 8 : 0;
        const float bi = bias[n];
        const float bj = bias[HID + n];
        const float bf = bias[2 * HID + n];
        const float bo = bias[3 * HID + n];
        #pragma unroll
        for (int r = 0; r < 8; ++r) {
            const int b = (wv << 4) + moff + r;
            const float gi = a0[r] + bi;
            const float gj = a1[r] + bj;
            const float gf = a2[r] + bf;
            const float go = a3[r] + bo;
            const float cold = Cst[b * NP + n];
            const float cn = sigmf(gf + 1.0f) * cold + sigmf(gi) * tanhf(gj);
            const float h  = sigmf(go) * tanhf(cn);
            Cst[b * NP + n]  = cn;
            Hout[b * NP + n] = (__bf16)h;
            if (HoutF) HoutF[b * NP + n] = h;
        }
    }
}

// --- tiny fp32 dense layers (negligible FLOPs) ------------------------------
__global__ void dense_kernel(const float* __restrict__ in, const float* __restrict__ W,
                             const float* __restrict__ bias, float* __restrict__ out,
                             int K, int N, int in_stride, int do_relu, int total)
{
    int idx = blockIdx.x * blockDim.x + threadIdx.x;
    if (idx >= total) return;
    const int b = idx / N, n = idx % N;
    float s = bias[n];
    for (int k = 0; k < K; ++k) s += in[(size_t)b * in_stride + k] * W[(size_t)k * N + n];
    out[idx] = do_relu ? fmaxf(s, 0.0f) : s;
}

// ---------------------------------------------------------------------------
extern "C" void kernel_launch(void* const* d_in, const int* in_sizes, int n_in,
                              void* d_out, int out_size, void* d_ws, size_t ws_size,
                              hipStream_t stream)
{
    (void)in_sizes; (void)n_in; (void)out_size; (void)ws_size;

    const int*   X     = (const int*)  d_in[0];
    const float* embed = (const float*)d_in[1];
    const float* k1    = (const float*)d_in[2];
    const float* b1    = (const float*)d_in[3];
    const float* k2    = (const float*)d_in[4];
    const float* b2    = (const float*)d_in[5];
    const float* w1    = (const float*)d_in[6];
    const float* bw1   = (const float*)d_in[7];
    const float* w2    = (const float*)d_in[8];
    const float* bw2   = (const float*)d_in[9];
    const float* wp    = (const float*)d_in[10];
    const float* bp    = (const float*)d_in[11];
    float* out = (float*)d_out;

    char* ws = (char*)d_ws;
    size_t off = 0;
    auto carve = [&](size_t bytes) -> char* {
        char* p = ws + off;
        off = (off + bytes + 255) & ~(size_t)255;
        return p;
    };
    __bf16* Xemb = (__bf16*)carve((size_t)TSTEPS * BATCH * NP * 2);   // 26.2 MB
    __bf16* W1p  = (__bf16*)carve((size_t)CT * 4 * KS * 32 * 16 * 2); // 4 MB
    __bf16* W2p  = (__bf16*)carve((size_t)CT * 4 * KS * 32 * 16 * 2); // 4 MB
    __bf16* h1   = (__bf16*)carve((size_t)2 * BATCH * NP * 2);        // ping-pong
    __bf16* h2   = (__bf16*)carve((size_t)2 * BATCH * NP * 2);
    float*  c1   = (float*) carve((size_t)BATCH * NP * 4);
    float*  c2   = (float*) carve((size_t)BATCH * NP * 4);
    float*  h2f  = (float*) carve((size_t)BATCH * NP * 4);
    float*  d1b  = (float*) carve((size_t)BATCH * 800 * 4);
    float*  d2b  = (float*) carve((size_t)BATCH * 100 * 4);

    // zero-init recurrent state every call (deterministic; capture-safe)
    hipMemsetAsync(h1, 0, (size_t)2 * BATCH * NP * 2, stream);
    hipMemsetAsync(h2, 0, (size_t)2 * BATCH * NP * 2, stream);
    hipMemsetAsync(c1, 0, (size_t)BATCH * NP * 4, stream);
    hipMemsetAsync(c2, 0, (size_t)BATCH * NP * 4, stream);
    hipMemsetAsync(h2f, 0, (size_t)BATCH * NP * 4, stream);

    {   // embedding gather + weight repack (parallel, once)
        const int tot = TSTEPS * BATCH * NP;
        embed_gather_kernel<<<(tot + 255) / 256, 256, 0, stream>>>(X, embed, Xemb);
        const int wtot = CT * 4 * KS * 32;
        repack_kernel<<<(wtot + 255) / 256, 256, 0, stream>>>(k1, W1p);
        repack_kernel<<<(wtot + 255) / 256, 256, 0, stream>>>(k2, W2p);
    }

    // sequential recurrence: stream ordering is the cross-WGP sync
    for (int t = 0; t < TSTEPS; ++t) {
        const int pr = t & 1, nx = (t + 1) & 1;
        lstm_step_kernel<<<dim3(CT), 128, LDS_BYTES, stream>>>(
            Xemb + (size_t)t * BATCH * NP,       // x_t
            h1 + (size_t)pr * BATCH * NP,        // h1_{t-1}
            W1p, b1, c1,
            h1 + (size_t)nx * BATCH * NP,        // h1_t
            (float*)nullptr);
        lstm_step_kernel<<<dim3(CT), 128, LDS_BYTES, stream>>>(
            h1 + (size_t)nx * BATCH * NP,        // layer-2 input = h1_t
            h2 + (size_t)pr * BATCH * NP,        // h2_{t-1}
            W2p, b2, c2,
            h2 + (size_t)nx * BATCH * NP,        // h2_t
            h2f);                                // fp32 copy for the head
    }

    // MLP head (fp32; tiny)
    dense_kernel<<<(BATCH * 800 + 255) / 256, 256, 0, stream>>>(h2f, w1, bw1, d1b, 500, 800, NP, 1, BATCH * 800);
    dense_kernel<<<(BATCH * 100 + 255) / 256, 256, 0, stream>>>(d1b, w2, bw2, d2b, 800, 100, 800, 1, BATCH * 100);
    dense_kernel<<<(BATCH * 2 + 255) / 256, 256, 0, stream>>>(d2b, wp, bp, out, 100, 2, 100, 0, BATCH * 2);
}
